// GPT2Attention_36438502539943
// MI455X (gfx1250) — matrix operand order repaired and checked by reading, not compiled, not run on hardware
//
#include <hip/hip_runtime.h>
#include <cstddef>
#include <cstdint>

// ---------------------------------------------------------------------------
// GPT-2 attention for MI455X (gfx1250, wave32, WMMA).
// Precision: bf16 operands / fp32 accumulate via v_wmma_f32_16x16x32_bf16.
// Bandwidth-bound on the 536MB attn_weights output -> weights written once.
// ---------------------------------------------------------------------------

typedef __attribute__((ext_vector_type(16))) __bf16 v16bf;
typedef __attribute__((ext_vector_type(8)))  float  v8f;

union Frag { v16bf v; unsigned u[8]; };
union Acc  { v8f   v; float    f[8]; };

__device__ __forceinline__ unsigned pack_bf16(float lo, float hi) {
    union { __bf16 b; unsigned short s; } a, c;
    a.b = (__bf16)lo; c.b = (__bf16)hi;
    return (unsigned)a.s | ((unsigned)c.s << 16);
}

// ---------------------------------------------------------------------------
// Generic GEMM: C[M,N] = A[M,K](f32) * B[K,N](f32) + bias[N]
// Output to fp32 (Cf) and/or bf16 (Cb). Tile 128x128, 256 threads = 8 waves,
// K-step 32 (one wmma K). LDS holds bf16 operands pre-packed as K-pairs so
// fragment loads read one K-pair per vgpr in the same contiguous k order (k = 16h + s) for A and B:
//   A-frag lane L (m=L%15? m=L&15, h=L>>4), vgpr v: K-pair index
//     pa = v + 8*h          (K = 2*pa, 2*pa+1)
//   B-frag lane L (n=L&15, h=L>>4), vgpr v: K-pair p = v + 8*h, column n
// ---------------------------------------------------------------------------
__global__ __launch_bounds__(256) void gemm_bias_wmma(
    const float* __restrict__ A, const float* __restrict__ B,
    const float* __restrict__ bias,
    float* __restrict__ Cf, __bf16* __restrict__ Cb,
    int M, int N, int K)
{
    __shared__ unsigned aP[128][17];   // [row][K-pair], padded vs bank conflicts
    __shared__ unsigned bP[16][128];   // [K-pair][col]

    const int tid  = threadIdx.x;
    const int wave = tid >> 5, lane = tid & 31;
    const int l16  = lane & 15, hh = lane >> 4;
    const int rowBase = blockIdx.y * 128;
    const int colBase = blockIdx.x * 128;

    Acc acc[8];
    #pragma unroll
    for (int n = 0; n < 8; ++n)
        #pragma unroll
        for (int j = 0; j < 8; ++j) acc[n].f[j] = 0.0f;

    // staging assignments
    const int arow = tid >> 1;            // 0..127
    const int apa0 = (tid & 1) * 8;       // pair base 0 or 8
    const int bcol = tid & 127;           // 0..127
    const int bp0  = (tid >> 7) * 8;      // pair base 0 or 8

    for (int kt = 0; kt < K; kt += 32) {
        __syncthreads();
        // ---- stage A tile 128x32 f32 -> bf16 pairs
        {
            const float* ap = A + (size_t)(rowBase + arow) * K + kt + 2 * apa0;
            #pragma unroll
            for (int j = 0; j < 8; ++j)
                aP[arow][apa0 + j] = pack_bf16(ap[2 * j], ap[2 * j + 1]);
        }
        // ---- stage B tile 32x128 f32 -> bf16 row-pairs
        #pragma unroll
        for (int j = 0; j < 8; ++j) {
            const int p = bp0 + j;
            const float* bp = B + (size_t)(kt + 2 * p) * N + colBase + bcol;
            bP[p][bcol] = pack_bf16(bp[0], bp[N]);
        }
        __syncthreads();

        // ---- A fragment for this wave's 16-row strip
        Frag af;
        #pragma unroll
        for (int v = 0; v < 8; ++v) {
            const int pa = v + 8 * hh;
            af.u[v] = aP[wave * 16 + l16][pa];
        }
        // ---- 8 column tiles
        #pragma unroll
        for (int n = 0; n < 8; ++n) {
            Frag bfr;
            #pragma unroll
            for (int v = 0; v < 8; ++v)
                bfr.u[v] = bP[v + 8 * hh][n * 16 + l16];
            acc[n].v = __builtin_amdgcn_wmma_f32_16x16x32_bf16(
                false, af.v, false, bfr.v, (short)0, acc[n].v, false, false);
        }
    }

    // ---- epilogue: C/D layout: lane -> col n*16+l16, vgpr v -> row v+8*hh
    #pragma unroll
    for (int n = 0; n < 8; ++n) {
        const int col = colBase + n * 16 + l16;
        const float bv = bias[col];
        #pragma unroll
        for (int v = 0; v < 8; ++v) {
            const int row = rowBase + wave * 16 + v + 8 * hh;
            const float val = acc[n].f[v] + bv;
            if (Cf) Cf[(size_t)row * N + col] = val;
            if (Cb) Cb[(size_t)row * N + col] = (__bf16)val;
        }
    }
}

// ---------------------------------------------------------------------------
// Attention: one workgroup per (b, h, 16-query-row tile).
// scores [16][2048] fp32 live entirely in LDS (128KB; CDNA5 allows 320KB/WG).
// qkv is bf16, layout [B*S][3D]: Q at col h*128, K at 2048+h*128, V at 4096+h*128.
// ---------------------------------------------------------------------------
constexpr int SEQ  = 2048;
constexpr int DMOD = 2048;
constexpr int NH   = 16;
constexpr int HD   = 128;
constexpr int TD   = 3 * DMOD;   // 6144

__global__ __launch_bounds__(256) void attn_wmma(
    const __bf16* __restrict__ qkv,
    float* __restrict__ wout,          // [B,H,S,S] normalized weights
    float* __restrict__ aout)          // [B,S,D] pre-projection attn output
{
    __shared__ float    sc[16][SEQ];      // 128 KB fp32 scores / exp values
    __shared__ unsigned vP[16][HD];       // 8 KB packed V K-row-pairs
    __shared__ float    red[16][16];
    __shared__ float    rowmax[16], rowinv[16];

    const int tid  = threadIdx.x;
    const int wave = tid >> 5, lane = tid & 31;
    const int l16  = lane & 15, hh = lane >> 4;
    const int qt = blockIdx.x, h = blockIdx.y, b = blockIdx.z;

    const unsigned*       qkvU = (const unsigned*)qkv;
    const unsigned short* qkvS = (const unsigned short*)qkv;

    // ---- load the 4 Q fragments (HD=128 = 4 x K32) for this 16-row tile
    Frag qf[4];
    {
        const int qrow = qt * 16 + l16;
        const size_t base = (size_t)(b * SEQ + qrow) * TD + h * HD; // even
        #pragma unroll
        for (int kc = 0; kc < 4; ++kc)
            #pragma unroll
            for (int v = 0; v < 8; ++v) {
                const int k0 = kc * 32 + 2 * v + hh * 16;
                qf[kc].u[v] = qkvU[(base + k0) >> 1];
            }
    }

    const float scale = 0.08838834764831845f;  // 1/sqrt(128)

    // ---- Phase 1: scores = scale * Q K^T   (wave w owns key tiles w, w+8, ...)
    for (int t = wave; t < SEQ / 16; t += 8) {
        const int kb = t * 16;
        Acc a;
        #pragma unroll
        for (int j = 0; j < 8; ++j) a.f[j] = 0.0f;
        // K row for this lane's B-frag column
        const size_t kbase = (size_t)(b * SEQ + kb + l16) * TD + DMOD + h * HD;
        #pragma unroll
        for (int kc = 0; kc < 4; ++kc) {
            Frag kf;   // B-frag: vgpr v holds d-rows (2v+16h, +1) -> contiguous pair
            #pragma unroll
            for (int v = 0; v < 8; ++v) {
                const int d0 = kc * 32 + 2 * v + 16 * hh;
                kf.u[v] = qkvU[(kbase + d0) >> 1];
            }
            a.v = __builtin_amdgcn_wmma_f32_16x16x32_bf16(
                false, qf[kc].v, false, kf.v, (short)0, a.v, false, false);
        }
        #pragma unroll
        for (int v = 0; v < 8; ++v)
            sc[v + 8 * hh][kb + l16] = a.f[v] * scale;
    }
    __syncthreads();

    // ---- Phase 2: softmax (16 threads per row, coalesced global writes)
    {
        const int r = tid >> 4, g = tid & 15;
        float m = -3.0e38f;
        for (int c = g; c < SEQ; c += 16) m = fmaxf(m, sc[r][c]);
        red[r][g] = m;
        __syncthreads();
        if (tid < 16) {
            float mm = red[tid][0];
            #pragma unroll
            for (int j = 1; j < 16; ++j) mm = fmaxf(mm, red[tid][j]);
            rowmax[tid] = mm;
        }
        __syncthreads();
        const float rm = rowmax[r];
        float sum = 0.0f;
        for (int c = g; c < SEQ; c += 16) {
            const float e = __expf(sc[r][c] - rm);
            sc[r][c] = e;            // keep unnormalized exp for P@V
            sum += e;
        }
        red[r][g] = sum;
        __syncthreads();
        if (tid < 16) {
            float ss = 0.0f;
            #pragma unroll
            for (int j = 0; j < 16; ++j) ss += red[tid][j];
            rowinv[tid] = 1.0f / ss;
        }
        __syncthreads();
        // single write of the 536MB weights tensor: normalized, coalesced
        const float inv = rowinv[r];
        float* wrow = wout + ((size_t)(b * NH + h) * SEQ + qt * 16 + r) * SEQ;
        for (int c = g; c < SEQ; c += 16) wrow[c] = sc[r][c] * inv;
    }

    // ---- Phase 3: O = (exp-scores @ V) * rowinv ; wave w owns d-cols [16w,16w+16)
    Acc o;
    #pragma unroll
    for (int j = 0; j < 8; ++j) o.f[j] = 0.0f;
    const int vc  = tid & 127;
    const int vp0 = (tid >> 7) * 8;
    for (int kc = 0; kc < SEQ / 32; ++kc) {
        const int kb = kc * 32;
        __syncthreads();
        // stage V tile 32x128 as packed K-row-pairs
        #pragma unroll
        for (int j = 0; j < 8; ++j) {
            const int p = vp0 + j;
            const size_t vi = (size_t)(b * SEQ + kb + 2 * p) * TD + 2 * DMOD + h * HD + vc;
            const unsigned lo = qkvS[vi];
            const unsigned hi = qkvS[vi + TD];
            vP[p][vc] = lo | (hi << 16);
        }
        __syncthreads();
        Frag pf, vfr;
        #pragma unroll
        for (int v = 0; v < 8; ++v) {
            const int k0 = kb + 2 * v + hh * 16;
            pf.u[v]  = pack_bf16(sc[l16][k0], sc[l16][k0 + 1]);  // P fp32->bf16 on the fly
            vfr.u[v] = vP[v + 8 * hh][wave * 16 + l16];
        }
        o.v = __builtin_amdgcn_wmma_f32_16x16x32_bf16(
            false, pf.v, false, vfr.v, (short)0, o.v, false, false);
    }
    #pragma unroll
    for (int v = 0; v < 8; ++v) {
        const int m = v + 8 * hh;
        const float val = o.f[v] * rowinv[m];
        aout[(size_t)(b * SEQ + qt * 16 + m) * DMOD + h * HD + wave * 16 + l16] = val;
    }
}

// ---------------------------------------------------------------------------
extern "C" void kernel_launch(void* const* d_in, const int* in_sizes, int n_in,
                              void* d_out, int out_size, void* d_ws, size_t ws_size,
                              hipStream_t stream) {
    (void)in_sizes; (void)n_in; (void)out_size; (void)ws_size;
    const float* hs     = (const float*)d_in[0];   // [2,2048,2048]
    const float* w_attn = (const float*)d_in[1];   // [2048,6144]
    const float* b_attn = (const float*)d_in[2];   // [6144]
    const float* w_proj = (const float*)d_in[3];   // [2048,2048]
    const float* b_proj = (const float*)d_in[4];   // [2048]

    float* out      = (float*)d_out;
    float* attn_out = out;                                   // [2,2048,2048]
    float* weights  = out + (size_t)2 * 2048 * 2048;         // [2,16,2048,2048]

    __bf16* qkv = (__bf16*)d_ws;                                       // 50.3 MB
    float*  aout = (float*)((char*)d_ws + (size_t)4096 * 6144 * 2);    // 33.5 MB

    const int M = 2 * SEQ;        // 4096
    dim3 blk(256);

    // 1) qkv(bf16) = hs @ w_attn + b_attn
    gemm_bias_wmma<<<dim3(TD / 128, M / 128), blk, 0, stream>>>(
        hs, w_attn, b_attn, nullptr, qkv, M, TD, DMOD);

    // 2) attention: weights (fp32, normalized) + pre-proj output (fp32)
    attn_wmma<<<dim3(SEQ / 16, NH, 2), blk, 0, stream>>>(qkv, weights, aout);

    // 3) attn_output = aout @ w_proj + b_proj
    gemm_bias_wmma<<<dim3(DMOD / 128, M / 128), blk, 0, stream>>>(
        aout, w_proj, b_proj, attn_out, nullptr, M, DMOD, DMOD);
}
